// ReformerLayer_21612275434066
// MI455X (gfx1250) — matrix operand order repaired and checked by reading, not compiled, not run on hardware
//
#include <hip/hip_runtime.h>
#include <hip/hip_bf16.h>
#include <math.h>

// ---------------- WMMA helpers (gfx1250, wave32) ----------------
typedef __attribute__((ext_vector_type(16))) _Float16 v16h;
typedef __attribute__((ext_vector_type(8)))  float    v8f;

#define BB   8
#define NN   1024
#define DD   512
#define HH   8
#define DH   64
#define BHN  (BB*HH)       // 64
#define NHR  4             // hash rounds
#define NBK  256           // buckets per round
#define NSORT 4096         // NHR * NN
#define NCHUNK 1024        // NSORT / BUCKET(4)

__device__ __forceinline__ v8f wmma16(const v16h& a, const v16h& b, const v8f& c) {
  // v_wmma_f32_16x16x32_f16  (neg_a, A, neg_b, B, c_mod, C, reuse_a, reuse_b)
  return __builtin_amdgcn_wmma_f32_16x16x32_f16(false, a, false, b, (short)0, c, false, false);
}

// A fragment: 16x32 f16. Lane L holds row M=L&15.
// L<16 : elems 0..7 = K 0..7,  elems 8..15 = K 16..23
// L>=16: elems 0..7 = K 8..15, elems 8..15 = K 24..31
__device__ __forceinline__ v16h load_a_frag(const float* __restrict__ Arow, int kk, int lane) {
  const int kbase = (lane < 16) ? 0 : 8;
  v16h a;
#pragma unroll
  for (int e = 0; e < 8; ++e) {
    a[e]     = (_Float16)Arow[kk + kbase + e];
    a[e + 8] = (_Float16)Arow[kk + 16 + kbase + e];
  }
  return a;
}

// C/D: VGPR i -> (lanes 0..15: M=i, N=lane) (lanes 16..31: M=8+i, N=lane-16)
__device__ __forceinline__ void store_c_tile(float* __restrict__ C, int ldc, int m0, int n0,
                                             int lane, const v8f& acc,
                                             const float* __restrict__ bias) {
  const int col = lane & 15;
  const int mo  = (lane < 16) ? 0 : 8;
  const float bv = bias ? bias[n0 + col] : 0.0f;
#pragma unroll
  for (int i = 0; i < 8; ++i)
    C[(size_t)(m0 + mo + i) * ldc + n0 + col] = acc[i] + bv;
}

// ------------- Kernel 0: pre-swizzle f32 weights -> f16 WMMA B-fragment layout --------
// Bsw[kt][tn][lane][e] , k = kt*32 + (e<8?0:16) + (lane<16?0:8) + (e&7) , n = tn*16 + (lane&15)
// => every B fragment becomes one aligned 32-byte load per lane.
__global__ void swizzle_b_kernel(const float* __restrict__ B, _Float16* __restrict__ Bsw,
                                 int K, int N) {
  const int idx = blockIdx.x * blockDim.x + threadIdx.x;
  if (idx >= K * N) return;
  const int e    = idx & 15;
  const int lane = (idx >> 4) & 31;
  const int r2   = idx >> 9;
  const int tn   = r2 % (N >> 4);
  const int kt   = r2 / (N >> 4);
  const int k = kt * 32 + ((e & 8) ? 16 : 0) + ((lane < 16) ? 0 : 8) + (e & 7);
  const int n = tn * 16 + (lane & 15);
  Bsw[idx] = (_Float16)B[(size_t)k * N + n];
}

// ---------------- Kernel 1/6: GEMM  C = A(MxK) * Bsw(KxN) [+bias] ---------------------
// One wave computes a 16x64 C strip: 1 A fragment feeds 4 WMMAs per k-step.
__global__ void gemm_wmma_kernel(const float* __restrict__ A, const _Float16* __restrict__ Bsw,
                                 float* __restrict__ C, const float* __restrict__ bias,
                                 int M, int N, int K) {
  const int lane = threadIdx.x & 31;
  const int wave = (blockIdx.x * blockDim.x + threadIdx.x) >> 5;
  const int strips_n = N >> 6;                    // 64-wide strips
  const int tm  = wave / strips_n;
  const int tn4 = wave - tm * strips_n;
  if (tm * 16 >= M) return;                       // wave-uniform
  const float* Arow = A + (size_t)(tm * 16 + (lane & 15)) * K;
  v8f acc0 = {}, acc1 = {}, acc2 = {}, acc3 = {};
  for (int kk = 0; kk < K; kk += 32) {
    __builtin_prefetch(Arow + kk + 64, 0, 1);     // global_prefetch_b8 (speculative-safe)
    const v16h af = load_a_frag(Arow, kk, lane);
    const _Float16* bp =
        Bsw + ((((size_t)(kk >> 5) * (N >> 4) + tn4 * 4) * 32) + lane) * 16;
    const v16h b0 = *(const v16h*)(bp + 0 * 512);
    const v16h b1 = *(const v16h*)(bp + 1 * 512);
    const v16h b2 = *(const v16h*)(bp + 2 * 512);
    const v16h b3 = *(const v16h*)(bp + 3 * 512);
    acc0 = wmma16(af, b0, acc0);
    acc1 = wmma16(af, b1, acc1);
    acc2 = wmma16(af, b2, acc2);
    acc3 = wmma16(af, b3, acc3);
  }
  store_c_tile(C, N, tm * 16, (tn4 * 4 + 0) * 16, lane, acc0, bias);
  store_c_tile(C, N, tm * 16, (tn4 * 4 + 1) * 16, lane, acc1, bias);
  store_c_tile(C, N, tm * 16, (tn4 * 4 + 2) * 16, lane, acc2, bias);
  store_c_tile(C, N, tm * 16, (tn4 * 4 + 3) * 16, lane, acc3, bias);
}

// ---------------- Kernel 2: LSH hashing (rotation GEMM via WMMA + argmax) --------------
// grid (64 bh, 64 token-tiles), 32 threads. Produces packed sort keys:
//   key = 1024*(bucket + 256*h) + t  (20 bits), packed = key*4096 + s_flat (fits u32)
__global__ void lsh_hash_kernel(const float* __restrict__ qk, const _Float16* __restrict__ rotsw,
                                unsigned* __restrict__ sortbuf) {
  const int bh   = blockIdx.x;
  const int tmj  = blockIdx.y;
  const int lane = threadIdx.x;
  const int b = bh >> 3, head = bh & 7;
  const int t = tmj * 16 + (lane & 15);
  const float* Arow = qk + ((size_t)(b * NN + t)) * DD + head * DH;
  __shared__ float tile[16][17];

  float maxv = -3.4e38f, minv = 3.4e38f;
  int maxi = 0, mini = 0;
  for (int ct = 0; ct < 32; ++ct) {             // 32 column tiles of 16 -> 512 cols
    v8f acc = {};
#pragma unroll
    for (int kk = 0; kk < DH; kk += 32) {
      const v16h af = load_a_frag(Arow, kk, lane);
      const v16h bf =
          *(const v16h*)(rotsw + ((((size_t)(kk >> 5) * 32 + ct) * 32) + lane) * 16);
      acc = wmma16(af, bf, acc);
    }
    const int col = lane & 15, mo = (lane < 16) ? 0 : 8;
#pragma unroll
    for (int i = 0; i < 8; ++i) tile[mo + i][col] = acc[i];
    __syncthreads();
    if (lane < 16) {
      const int h = ct >> 3;
      const int ibase = ct * 16 - h * 128;      // index within this hash round
      for (int c2 = 0; c2 < 16; ++c2) {
        const float v = tile[lane][c2];
        const int idx = ibase + c2;
        if (v > maxv) { maxv = v; maxi = idx; }   // first-occurrence argmax
        if (v < minv) { minv = v; mini = idx; }   // first-occurrence argmin
      }
      if ((ct & 7) == 7) {                      // round h complete
        const int bucket = (maxv >= -minv) ? maxi : (128 + mini);
        const int tt = tmj * 16 + lane;
        const unsigned s_flat = (unsigned)(h * NN + tt);
        const unsigned key = (unsigned)(NN * (bucket + h * NBK) + tt);
        sortbuf[(size_t)bh * NSORT + s_flat] = (key << 12) | s_flat;
        maxv = -3.4e38f; minv = 3.4e38f; maxi = 0; mini = 0;
      }
    }
    __syncthreads();
  }
}

// ---------------- Kernel 3: bitonic sort of 4096 packed keys per bh --------------------
__global__ void bitonic_sort_kernel(unsigned* __restrict__ sortbuf) {
  __shared__ unsigned s[NSORT];
  unsigned* g = sortbuf + (size_t)blockIdx.x * NSORT;
  for (int i = threadIdx.x; i < NSORT; i += blockDim.x) s[i] = g[i];
  __syncthreads();
  for (int k = 2; k <= NSORT; k <<= 1) {
    for (int j = k >> 1; j > 0; j >>= 1) {
      for (int i = threadIdx.x; i < NSORT; i += blockDim.x) {
        const int ixj = i ^ j;
        if (ixj > i) {
          const unsigned a = s[i], c = s[ixj];
          const bool up = ((i & k) == 0);
          if (up ? (a > c) : (a < c)) { s[i] = c; s[ixj] = a; }
        }
      }
      __syncthreads();
    }
  }
  for (int i = threadIdx.x; i < NSORT; i += blockDim.x) g[i] = s[i];
}

// ---------------- Kernel 4: chunked attention (4 q x 8 kv), scatter outputs ------------
__global__ void lsh_attn_kernel(const float* __restrict__ qk, const float* __restrict__ vv,
                                const unsigned* __restrict__ sortbuf,
                                float* __restrict__ o_full, float* __restrict__ slog) {
  const int w    = threadIdx.x >> 5;          // wave in block (8 waves)
  const int lane = threadIdx.x & 31;
  const int gwave = blockIdx.x * 8 + w;
  const int bh = gwave >> 10;                 // 1024 chunks per bh
  const int c  = gwave & 1023;
  const int b = bh >> 3, head = bh & 7;

  __shared__ float    qs[8][4][DH];
  __shared__ float    ks[8][8][DH];
  __shared__ float    vs[8][8][DH];
  __shared__ float    inrm[8][8];
  __shared__ int      tq[8][4];
  __shared__ int      tk[8][8];
  __shared__ unsigned sfl[8][4];
  __shared__ float    pr[8][4][8];
  __shared__ float    ls[8][4];

  const unsigned* sb = sortbuf + (size_t)bh * NSORT;
  if (lane < 8) {                              // 8 kv rows: 0..3 this chunk, 4..7 prev
    const int pos = (lane < 4) ? (c * 4 + lane)
                               : (((c + NCHUNK - 1) & (NCHUNK - 1)) * 4 + (lane - 4));
    const unsigned s = sb[pos] & 0xFFFu;
    const int t = (int)(s & (NN - 1));
    tk[w][lane] = t;
    if (lane < 4) { tq[w][lane] = t; sfl[w][lane] = s; }
  }
  __syncthreads();

  // stage q rows (=qk), k rows (=qk, to be normalized), v rows
  for (int e = lane; e < 20 * DH; e += 32) {
    const int r = e >> 6, f = e & (DH - 1);
    if (r < 4) {
      qs[w][r][f] = qk[((size_t)(b * NN + tq[w][r])) * DD + head * DH + f];
    } else if (r < 12) {
      const int j = r - 4;
      ks[w][j][f] = qk[((size_t)(b * NN + tk[w][j])) * DD + head * DH + f];
    } else {
      const int j = r - 12;
      vs[w][j][f] = vv[((size_t)(b * NN + tk[w][j])) * DD + head * DH + f];
    }
  }
  __syncthreads();
  if (lane < 8) {                              // unit-length keys: 1/max(||k||,1e-6)
    float ss = 0.0f;
    for (int f = 0; f < DH; ++f) { const float x = ks[w][lane][f]; ss += x * x; }
    inrm[w][lane] = 1.0f / fmaxf(sqrtf(ss), 1e-6f);
  }
  __syncthreads();

  const int qi = lane >> 3, kj = lane & 7;     // 32 lanes = 4x8 score matrix
  float dot = 0.0f;
  for (int f = 0; f < DH; ++f) dot += qs[w][qi][f] * ks[w][kj][f];
  dot *= inrm[w][kj] * 0.125f;                 // dh^-0.5 = 1/8
  if (tq[w][qi] == tk[w][kj]) dot = -5e4f;     // TOKEN_SELF_ATTN_VALUE
  float m = dot;
  for (int off = 1; off < 8; off <<= 1) m = fmaxf(m, __shfl_xor(m, off, 8));
  float ex = __expf(dot - m), se = ex;
  for (int off = 1; off < 8; off <<= 1) se += __shfl_xor(se, off, 8);
  const float lse = m + __logf(se);
  pr[w][qi][kj] = __expf(dot - lse);
  if (kj == 0) ls[w][qi] = lse;
  __syncthreads();

  // bo = probs @ bv ; scatter by original flat index (this IS the undo_sort)
  const int oi = lane >> 3, f0 = (lane & 7) * 8;
  float out[8];
#pragma unroll
  for (int d = 0; d < 8; ++d) {
    float a = 0.0f;
#pragma unroll
    for (int j = 0; j < 8; ++j) a += pr[w][oi][j] * vs[w][j][f0 + d];
    out[d] = a;
  }
  const unsigned s = sfl[w][oi];
  float* orow = o_full + ((size_t)bh * NSORT + s) * DH + f0;
#pragma unroll
  for (int d = 0; d < 8; ++d) orow[d] = out[d];
  if (kj == 0) slog[(size_t)bh * NSORT + s] = ls[w][oi];
}

// ---------------- Kernel 5: combine hash rounds (softmax over round LSEs) --------------
__global__ void combine_kernel(const float* __restrict__ o_full, const float* __restrict__ slog,
                               float* __restrict__ ctx) {
  const int e = blockIdx.x * blockDim.x + threadIdx.x;     // BHN*NN*DH = 4M
  if (e >= BHN * NN * DH) return;
  const int f = e & (DH - 1);
  const int t = (e >> 6) & (NN - 1);
  const int bh = e >> 16;
  const float* lg = slog + (size_t)bh * NSORT + t;
  const float l0 = lg[0], l1 = lg[NN], l2 = lg[2 * NN], l3 = lg[3 * NN];
  const float m = fmaxf(fmaxf(l0, l1), fmaxf(l2, l3));
  const float w0 = __expf(l0 - m), w1 = __expf(l1 - m), w2 = __expf(l2 - m), w3 = __expf(l3 - m);
  const float inv = 1.0f / (w0 + w1 + w2 + w3);
  const float* ob = o_full + ((size_t)bh * NSORT + t) * DH + f;
  const float val = (w0 * ob[0] + w1 * ob[(size_t)NN * DH] +
                     w2 * ob[(size_t)2 * NN * DH] + w3 * ob[(size_t)3 * NN * DH]) * inv;
  const int b = bh >> 3, head = bh & 7;
  ctx[((size_t)(b * NN + t)) * DD + head * DH + f] = val;
}

// ---------------- host-side orchestration ----------------
extern "C" void kernel_launch(void* const* d_in, const int* in_sizes, int n_in,
                              void* d_out, int out_size, void* d_ws, size_t ws_size,
                              hipStream_t stream) {
  const float* queries = (const float*)d_in[0];
  const float* Wqk     = (const float*)d_in[3];
  const float* Wv      = (const float*)d_in[4];
  const float* Wout    = (const float*)d_in[5];
  const float* b_out   = (const float*)d_in[6];
  const float* rot     = (const float*)d_in[7];
  (void)in_sizes; (void)n_in; (void)out_size; (void)ws_size;

  char* ws = (char*)d_ws;
  const size_t MB = 1024u * 1024u;
  float*    qkbuf   = (float*)(ws);                     // 16 MB  (8192 x 512 f32)
  float*    vbuf    = (float*)(ws + 16 * MB);           // 16 MB
  unsigned* sortbuf = (unsigned*)(ws + 32 * MB);        //  1 MB  (64 x 4096 u32)
  float*    o_full  = (float*)(ws + 33 * MB);           // 64 MB  (64 x 4096 x 64 f32)
  float*    slog    = (float*)(ws + 97 * MB);           //  1 MB
  float*    ctx     = (float*)(ws + 98 * MB);           // 16 MB
  _Float16* wqk16   = (_Float16*)(ws + 114 * MB);       // 0.5 MB (swizzled f16)
  _Float16* wv16    = (_Float16*)(ws + 115 * MB);       // 0.5 MB
  _Float16* wout16  = (_Float16*)(ws + 116 * MB);       // 0.5 MB
  _Float16* rot16   = (_Float16*)(ws + 117 * MB);       // 64 KB
  float*    outp    = (float*)d_out;

  const int M = BB * NN;                                // 8192

  // 0) pre-swizzle weights + rotations into WMMA B-fragment f16 layout
  swizzle_b_kernel<<<(DD * DD) / 256, 256, 0, stream>>>(Wqk,  wqk16,  DD, DD);
  swizzle_b_kernel<<<(DD * DD) / 256, 256, 0, stream>>>(Wv,   wv16,   DD, DD);
  swizzle_b_kernel<<<(DD * DD) / 256, 256, 0, stream>>>(Wout, wout16, DD, DD);
  swizzle_b_kernel<<<(DH * DD) / 256, 256, 0, stream>>>(rot,  rot16,  DH, DD);

  // waves = (8192/16)*(512/64) = 4096 ; 8 waves per 256-thread block
  const int gemm_blocks = (M / 16) * (DD / 64) / 8;

  // 1) qk = Q @ Wqk ; 2) v = Q @ Wv     (WMMA f16 -> f32, 4 wmma per k-step)
  gemm_wmma_kernel<<<gemm_blocks, 256, 0, stream>>>(queries, wqk16, qkbuf, nullptr, M, DD, DD);
  gemm_wmma_kernel<<<gemm_blocks, 256, 0, stream>>>(queries, wv16,  vbuf,  nullptr, M, DD, DD);

  // 3) LSH hashing: per-head rotation GEMM + argmax -> packed sort keys
  lsh_hash_kernel<<<dim3(BHN, NN / 16), 32, 0, stream>>>(qkbuf, rot16, sortbuf);

  // 4) stable sort (unique 32-bit packed keys) per head-batch
  bitonic_sort_kernel<<<BHN, 1024, 0, stream>>>(sortbuf);

  // 5) chunked attention, scatter to original order
  lsh_attn_kernel<<<(BHN * NCHUNK) / 8, 256, 0, stream>>>(qkbuf, vbuf, sortbuf, o_full, slog);

  // 6) combine hash rounds into head-interleaved context
  combine_kernel<<<(BHN * NN * DH) / 256, 256, 0, stream>>>(o_full, slog, ctx);

  // 7) out = ctx @ Wout + b_out
  gemm_wmma_kernel<<<gemm_blocks, 256, 0, stream>>>(ctx, wout16, outp, b_out, M, DD, DD);
}
